// LegendrePolynomials_66185446032208
// MI455X (gfx1250) — matrix operand order repaired and checked by reading, not compiled, not run on hardware
//
#include <hip/hip_runtime.h>

// ---------------------------------------------------------------------------
// LegendrePolynomials fused GEMM for gfx1250 (MI455X, wave32, WMMA)
//   y[b,o] = sum_{i,d} P_d(tanh(x[b,i])) * coeffs[i,o,d]
// v5: compile-time degree pipeline (no runtime state machine -> no spills),
//     async global->LDS DMA for B tiles staged one step ahead,
//     double-buffered LDS, single barrier per K-step.
// ---------------------------------------------------------------------------

typedef _Float16 h4   __attribute__((ext_vector_type(4)));
typedef _Float16 h8   __attribute__((ext_vector_type(8)));
typedef _Float16 v16h __attribute__((ext_vector_type(16)));
typedef float    v8f  __attribute__((ext_vector_type(8)));

#define IN_DIM  512
#define OUT_DIM 512
#define NDEG    9                // DEGREE + 1
#define NIC     (IN_DIM / BK)    // 16 i-chunks
#define BM      64               // rows per block
#define BN      512              // full output width per block
#define BK      32               // K-slab per step
#define BKP     40               // padded LDS row pitch (halves)
#define ABYTES  (BM * BKP * 2)   // 5120
#define BBYTES  (BN * BKP * 2)   // 40960

// Prepass: Bs[ic][d][o][k] = (f16) coeffs[ic*32+k][o][d]
// -> every (ic,d) step's 512x32 B tile is one contiguous 32KB slab.
__global__ __launch_bounds__(256) void prep_coeffs(const float* __restrict__ coeffs,
                                                   _Float16* __restrict__ bs) {
  int t = blockIdx.x * 256 + threadIdx.x;   // 0 .. 512*512-1
  int i = t & (IN_DIM - 1);
  int o = t >> 9;
  const int ic = i >> 5;
  const int il = i & 31;
  const float* src = coeffs + ((size_t)i * OUT_DIM + o) * NDEG;
#pragma unroll
  for (int d = 0; d < NDEG; ++d) {
    bs[(((size_t)ic * NDEG + d) * OUT_DIM + o) * BK + il] = (_Float16)src[d];
  }
}

__global__ __launch_bounds__(512) void legendre_wmma(const float* __restrict__ x,
                                                     const _Float16* __restrict__ bs,
                                                     float* __restrict__ out) {
  __shared__ __align__(16) _Float16 ldsA[2][BM * BKP];   // 2 x 5 KB
  __shared__ __align__(16) _Float16 ldsB[2][BN * BKP];   // 2 x 40 KB

  const int tid  = threadIdx.x;            // 0..511, 16 wave32s
  const int row0 = blockIdx.x * BM;

  // --- A-producer indexing: thread owns 1 row x 4 cols of the 64x32 tile ----
  const int arow = tid >> 3;               // 0..63
  const int acg  = (tid & 7) << 2;         // 0,4,...,28

  // --- wave/WMMA indexing: 16 waves = 2(m) x 8(n); wave tile 32 x 64 -------
  const int lane = tid & 31;
  const int wid  = tid >> 5;               // 0..15
  const int wm   = (wid & 1) * 32;
  const int wn   = (wid >> 1) * 64;
  const int lcol = lane & 15;
  const int lhi  = lane >> 4;

  const uint32_t ldsB_base = (uint32_t)(uintptr_t)&ldsB[0][0];

  const v8f vzero = {0.f, 0.f, 0.f, 0.f, 0.f, 0.f, 0.f, 0.f};
  v8f acc[2][4];                           // 2(tm) x 4(tn): 64 acc VGPRs
#pragma unroll
  for (int tm = 0; tm < 2; ++tm)
#pragma unroll
    for (int tn = 0; tn < 4; ++tn)
      acc[tm][tn] = vzero;

  // Producer recurrence state (runs exactly one step ahead of the consume).
  float4 xf, pm1, pm2;

  // ---- async DMA of one 32KB contiguous B slab into ldsB[nb] --------------
  auto stageB = [&](int t, int nb) {
    uint64_t ga = (uint64_t)(uintptr_t)(bs + (size_t)t * (OUT_DIM * BK)) +
                  (uint32_t)(wid * 2048 + lane * 16);
    uint32_t la = ldsB_base + nb * BBYTES +
                  (uint32_t)((wid * 32 + (lane >> 2)) * (BKP * 2) +
                             ((lane & 3) << 4));
#pragma unroll
    for (int j = 0; j < 4; ++j) {
      asm volatile("global_load_async_to_lds_b128 %0, %1, off"
                   :: "v"(la), "v"(ga) : "memory");
      la += BKP * 2 * 8;   // 8 rows of padded pitch
      ga += 512;
    }
  };

  auto storePd = [&](float4 pd, int nb) {
    h4 hv;
    hv[0] = (_Float16)pd.x; hv[1] = (_Float16)pd.y;
    hv[2] = (_Float16)pd.z; hv[3] = (_Float16)pd.w;
    *(h4*)&ldsA[nb][arow * BKP + acg] = hv;
  };

  // Degree-0 producer for i-chunk icn: load x slab, tanh, P0 = 1.
  auto produce0 = [&](int icn, int nb) {
    const float4 xv =
        *(const float4*)(x + (size_t)(row0 + arow) * IN_DIM + icn * BK + acg);
    xf.x = tanhf(xv.x); xf.y = tanhf(xv.y);
    xf.z = tanhf(xv.z); xf.w = tanhf(xv.w);
    pm2 = make_float4(1.f, 1.f, 1.f, 1.f);
    storePd(pm2, nb);
  };

  // ---- prologue: stage step (ic=0, d=0) into buffer 0 ---------------------
  stageB(0, 0);
  produce0(0, 0);

  for (int ic = 0; ic < NIC; ++ic) {
#pragma unroll
    for (int d = 0; d < NDEG; ++d) {
      const int cbuf = (ic + d) & 1;       // s = ic*9 + d; 9 odd -> (ic+d)&1
      const int nbuf = cbuf ^ 1;

      // Drain this wave's async DMA, then barrier: cbuf is now valid.
      asm volatile("s_wait_asynccnt 0x0" ::: "memory");
      __syncthreads();

      // Stage the *next* step first so the DMA + producer VALU overlap the
      // WMMA phase below (writes go to nbuf; its readers finished before the
      // barrier above).
      if (d + 1 < NDEG) {
        stageB(ic * NDEG + d + 1, nbuf);
        float4 pd;
        if (d + 1 == 1) {
          pd = xf;                                     // P1 = xf
          pm1 = pd;
        } else {
          const float c1 = (2.f * (d + 1) - 1.f) / (float)(d + 1);
          const float c2 = ((float)(d + 1) - 1.f) / (float)(d + 1);
          pd.x = c1 * xf.x * pm1.x - c2 * pm2.x;
          pd.y = c1 * xf.y * pm1.y - c2 * pm2.y;
          pd.z = c1 * xf.z * pm1.z - c2 * pm2.z;
          pd.w = c1 * xf.w * pm1.w - c2 * pm2.w;
          pm2 = pm1; pm1 = pd;
        }
        storePd(pd, nbuf);
      } else if (ic + 1 < NIC) {
        stageB((ic + 1) * NDEG, nbuf);
        produce0(ic + 1, nbuf);
      }

      // ---- consume step (ic,d): A fragments (16-bit A 16x32 layout) -------
      v16h af[2];
#pragma unroll
      for (int tm = 0; tm < 2; ++tm) {
        const int r = wm + tm * 16 + lcol;
        const h8 lo = *(const h8*)&ldsA[cbuf][r * BKP + lhi * 16];
        const h8 hi = *(const h8*)&ldsA[cbuf][r * BKP + lhi * 16 + 8];
        af[tm] = __builtin_shufflevector(lo, hi, 0, 1, 2, 3, 4, 5, 6, 7,
                                         8, 9, 10, 11, 12, 13, 14, 15);
      }

      // ---- B fragments (register double-buffer), 8 WMMAs per K-step -------
      v16h bfr[2];
      {
        const int c0 = wn + lcol;
        const h8 blo = *(const h8*)&ldsB[cbuf][c0 * BKP + lhi * 16];
        const h8 bhi = *(const h8*)&ldsB[cbuf][c0 * BKP + lhi * 16 + 8];
        bfr[0] = __builtin_shufflevector(blo, bhi, 0, 1, 2, 3, 4, 5, 6, 7,
                                         8, 9, 10, 11, 12, 13, 14, 15);
      }
#pragma unroll
      for (int tn = 0; tn < 4; ++tn) {
        if (tn + 1 < 4) {
          const int c = wn + (tn + 1) * 16 + lcol;
          const h8 blo = *(const h8*)&ldsB[cbuf][c * BKP + lhi * 16];
          const h8 bhi = *(const h8*)&ldsB[cbuf][c * BKP + lhi * 16 + 8];
          bfr[(tn + 1) & 1] = __builtin_shufflevector(
              blo, bhi, 0, 1, 2, 3, 4, 5, 6, 7, 8, 9, 10, 11, 12, 13, 14, 15);
        }
        acc[0][tn] = __builtin_amdgcn_wmma_f32_16x16x32_f16(
            false, af[0], false, bfr[tn & 1], (short)0, acc[0][tn], false, false);
        acc[1][tn] = __builtin_amdgcn_wmma_f32_16x16x32_f16(
            false, af[1], false, bfr[tn & 1], (short)0, acc[1][tn], false, false);
      }
    }
  }

  // ---- epilogue: C layout -> lane holds column lcol, rows in the 8 VGPRs ----
#pragma unroll
  for (int tm = 0; tm < 2; ++tm) {
#pragma unroll
    for (int tn = 0; tn < 4; ++tn) {
      const int c = wn + tn * 16 + lcol;
#pragma unroll
      for (int r = 0; r < 8; ++r) {
        const int rr = row0 + wm + tm * 16 + lhi * 8 + r;
        out[(size_t)rr * OUT_DIM + c] = acc[tm][tn][r];
      }
    }
  }
}

extern "C" void kernel_launch(void* const* d_in, const int* in_sizes, int n_in,
                              void* d_out, int out_size, void* d_ws, size_t ws_size,
                              hipStream_t stream) {
  const float* x      = (const float*)d_in[0];   // (16,1024,512) f32
  const float* coeffs = (const float*)d_in[1];   // (512,512,9)   f32
  float*       out    = (float*)d_out;           // (16,1024,512) f32
  _Float16*    bs     = (_Float16*)d_ws;         // needs 9*512*512*2 = 4.5 MB

  const int rows = in_sizes[0] / IN_DIM;         // 16384

  prep_coeffs<<<(IN_DIM * OUT_DIM) / 256, 256, 0, stream>>>(coeffs, bs);

  legendre_wmma<<<rows / BM, 512, 0, stream>>>(x, bs, out);
}
